// TransformerEncoderLayerBase_67018669686849
// MI455X (gfx1250) — compile-verified, matrix operand order checked by reading, not run on hardware
//
#include <hip/hip_runtime.h>
#include <hip/hip_bf16.h>

// ---------------- problem constants ----------------
#define S_LEN   1024
#define BATCH   2
#define DMODEL  1024
#define NHEAD   16
#define HDIM    64
#define EA_N    4
#define EF_N    8
#define FF_DIM  2048
#define NTOK    (S_LEN*BATCH)       // 2048 tokens
#define EPSL    1e-5f

// ---------------- CDNA5 WMMA types ----------------
typedef __attribute__((ext_vector_type(16))) __bf16 v16bf;
typedef __attribute__((ext_vector_type(8)))  float  v8f;

union BF16Frag {
    v16bf v;
    unsigned short u[16];
    unsigned int   w32[8];
};

// fp32 -> bf16 round-to-nearest-even
__device__ __forceinline__ unsigned short f2bf(float f) {
    unsigned int u = __float_as_uint(f);
    u = (u + 0x7FFFu + ((u >> 16) & 1u)) >> 16;
    return (unsigned short)u;
}
__device__ __forceinline__ float bf2f(unsigned short h) {
    return __uint_as_float(((unsigned int)h) << 16);
}

// ---------------- gfx1250 async global->LDS staging (guarded, probe-confirmed) ----------------
#if defined(__has_builtin)
#if __has_builtin(__builtin_amdgcn_global_load_async_to_lds_b128)
#define HAVE_ASYNC_B128 1
#endif
#if __has_builtin(__builtin_amdgcn_s_wait_asynccnt)
#define HAVE_WAIT_ASYNC 1
#endif
#endif

typedef int v4i_as __attribute__((vector_size(16)));
typedef __attribute__((address_space(1))) v4i_as* gvec_ptr;
typedef __attribute__((address_space(3))) v4i_as* lvec_ptr;

__device__ __forceinline__ void ld_g2l_b128(void* lds, const void* g) {
#if HAVE_ASYNC_B128
    __builtin_amdgcn_global_load_async_to_lds_b128((gvec_ptr)g, (lvec_ptr)lds, 0, 0);
#else
    *(uint4*)lds = *(const uint4*)g;
#endif
}
__device__ __forceinline__ void wait_async_loads() {
#if HAVE_ASYNC_B128
#if HAVE_WAIT_ASYNC
    __builtin_amdgcn_s_wait_asynccnt(0);
#else
    asm volatile("s_wait_asynccnt 0" ::: "memory");
#endif
#endif
}

// Load a 16x32 bf16 A-style fragment from LDS (row-major [row][k], rowStride in elems).
// CDNA5 16-bit A layout: lanes 0-15 rows M, K = {0..7,16..23}; lanes 16-31 K = {8..15,24..31}.
// Consecutive-K pairs per VGPR -> compiler merges to ds_load_b128 pairs.
__device__ __forceinline__ void load_frag_rm(BF16Frag& f, const unsigned short* base,
                                             int rowStride, int row, int k0, int lane) {
    const unsigned short* p = base + row * rowStride + k0;
    int kb = (lane & 16) ? 8 : 0;
#pragma unroll
    for (int q = 0; q < 8; q++) {
        int k = kb + ((q & 3) << 1) + ((q >> 2) << 4);
        f.w32[q] = *(const unsigned int*)(p + k);
    }
}

// ---------------- utility kernels ----------------
__global__ __launch_bounds__(256) void copy_f4_kernel(const float4* __restrict__ src,
                                                      float4* __restrict__ dst, int n4) {
    int i = blockIdx.x * 256 + threadIdx.x;
    if (i < n4) dst[i] = src[i];
}

__global__ void zero_aux_kernel(float* p) {
    if (threadIdx.x < 32) p[threadIdx.x] = 0.f;
}

// one-time f32 [K,N] -> bf16 transposed [N,K] (64x64 LDS tiles, coalesced both ways)
__global__ __launch_bounds__(256)
void convT_bf16_kernel(const float* __restrict__ src, unsigned short* __restrict__ dst,
                       int K, int N) {
    __shared__ unsigned short Ts[64][72];   // 144B row stride: 16B aligned
    int k0 = blockIdx.x * 64, n0 = blockIdx.y * 64;
    int tid = threadIdx.x;
#pragma unroll
    for (int j = 0; j < 4; j++) {
        int i = tid + j * 256;          // 1024 float4 = 64x64 f32
        int r = i >> 4, c4 = i & 15;    // r: k-row, c4*4: n-col
        float4 f = *(const float4*)(src + (size_t)(k0 + r) * N + n0 + c4 * 4);
        Ts[c4 * 4 + 0][r] = f2bf(f.x);
        Ts[c4 * 4 + 1][r] = f2bf(f.y);
        Ts[c4 * 4 + 2][r] = f2bf(f.z);
        Ts[c4 * 4 + 3][r] = f2bf(f.w);
    }
    __syncthreads();
#pragma unroll
    for (int j = 0; j < 2; j++) {
        int i = tid + j * 256;          // 512 uint4 = 64x64 bf16
        int rn = i >> 3, seg = i & 7;
        *(uint4*)(dst + (size_t)(n0 + rn) * K + k0 + seg * 8) =
            *(const uint4*)&Ts[rn][seg * 8];
    }
}

// ---------------- LayerNorm -> bf16 ----------------
__global__ __launch_bounds__(256)
void ln_bf16_kernel(const float* __restrict__ x, const float* __restrict__ sc,
                    const float* __restrict__ bi, unsigned short* __restrict__ out) {
    __shared__ float red[16];
    int tid = threadIdx.x, lane = tid & 31, w = tid >> 5;
    const float* row = x + (size_t)blockIdx.x * DMODEL;
    float v[4];
    float s = 0.f, ss = 0.f;
#pragma unroll
    for (int i = 0; i < 4; i++) {
        v[i] = row[tid + i * 256];
        s += v[i]; ss += v[i] * v[i];
    }
#pragma unroll
    for (int o = 16; o >= 1; o >>= 1) {
        s  += __shfl_xor(s,  o, 32);
        ss += __shfl_xor(ss, o, 32);
    }
    if (lane == 0) { red[w] = s; red[8 + w] = ss; }
    __syncthreads();
    float ts = 0.f, tss = 0.f;
#pragma unroll
    for (int ww = 0; ww < 8; ww++) { ts += red[ww]; tss += red[8 + ww]; }
    float mu  = ts * (1.f / DMODEL);
    float var = tss * (1.f / DMODEL) - mu * mu;
    float r   = rsqrtf(var + EPSL);
    unsigned short* orow = out + (size_t)blockIdx.x * DMODEL;
#pragma unroll
    for (int i = 0; i < 4; i++) {
        int d = tid + i * 256;
        orow[d] = f2bf((v[i] - mu) * r * sc[d] + bi[d]);
    }
}

// ---------------- gating (one wave per token, top-2 of E) ----------------
template <int E>
__global__ __launch_bounds__(256)
void gate_kernel(const unsigned short* __restrict__ hbf, const float* __restrict__ Wg,
                 const unsigned char* __restrict__ halt,
                 float* __restrict__ gates,   // [E][NTOK] expert-major
                 float* __restrict__ imp, float* __restrict__ loadc, float* __restrict__ zsum) {
    int lane = threadIdx.x & 31, w = threadIdx.x >> 5;
    int token = blockIdx.x * 8 + w;
    float le[E];
#pragma unroll
    for (int e = 0; e < E; e++) le[e] = 0.f;
    const unsigned short* hrow = hbf + (size_t)token * DMODEL;
    for (int d = lane; d < DMODEL; d += 32) {
        float hv = bf2f(hrow[d]);
#pragma unroll
        for (int e = 0; e < E; e++) le[e] += hv * Wg[d * E + e];
    }
#pragma unroll
    for (int e = 0; e < E; e++)
#pragma unroll
        for (int o = 16; o >= 1; o >>= 1) le[e] += __shfl_xor(le[e], o, 32);

    if (lane == 0) {
        float m = le[0];
#pragma unroll
        for (int e = 1; e < E; e++) m = fmaxf(m, le[e]);
        float pe[E], psum = 0.f;
#pragma unroll
        for (int e = 0; e < E; e++) { pe[e] = __expf(le[e] - m); psum += pe[e]; }
        float inv = 1.f / psum;
        int i1 = 0;
#pragma unroll
        for (int e = 1; e < E; e++) if (pe[e] > pe[i1]) i1 = e;
        int i2 = (i1 == 0) ? 1 : 0;
#pragma unroll
        for (int e = 0; e < E; e++) if (e != i1 && pe[e] > pe[i2]) i2 = e;
        float v1 = pe[i1] * inv, v2 = pe[i2] * inv;
        float denom = v1 + v2 + 1e-9f;
        float keep = halt[token] ? 0.f : 1.f;
        float g1 = v1 / denom * keep, g2 = v2 / denom * keep;
        float lse = m + __logf(psum);
        atomicAdd(zsum, lse * lse);
#pragma unroll
        for (int e = 0; e < E; e++) {
            float pr = pe[e] * inv;
            float g = (e == i1) ? g1 : ((e == i2) ? g2 : 0.f);
            gates[(size_t)e * NTOK + token] = g;
            atomicAdd(imp + e, pr);
            atomicAdd(loadc + e, (g > 0.f) ? 1.f : 0.f);
        }
    }
}

// ---------------- bf16 WMMA GEMM ----------------
// C[M,N] = op(A[M,K](bf16) @ Bt[N,K](bf16,transposed) + bias) ; M,N from grid (x128).
// Tile 128x128x64; 8 waves, wave tile 64x32 (4x2 16x16 frags), 16 wmma / K-tile.
// transT: scatter bf16 output as [(token%B)*D + col][token/B]  (V for attention)
__global__ __launch_bounds__(256)
void gemm_bf16_kernel(const unsigned short* __restrict__ A, int lda,
                      const unsigned short* __restrict__ Bt, int ldb,
                      const float* __restrict__ bias,
                      const float* __restrict__ rowScale,
                      unsigned short* __restrict__ Cbf,
                      float* __restrict__ Cacc, int ldc,
                      int K, int relu, int transT) {
    __shared__ unsigned short As[128 * 64];   // [m][k]
    __shared__ unsigned short Bs[128 * 64];   // [n][k]
    int tid = threadIdx.x, lane = tid & 31, wid = tid >> 5;
    int mBlk = blockIdx.y * 128, nBlk = blockIdx.x * 128;
    int mW = (wid & 1) * 64, nW = (wid >> 1) * 32;

    v8f acc[4][2];
#pragma unroll
    for (int im = 0; im < 4; im++)
#pragma unroll
        for (int in = 0; in < 2; in++)
#pragma unroll
            for (int i = 0; i < 8; i++) acc[im][in][i] = 0.f;

    for (int k0 = 0; k0 < K; k0 += 64) {
        __syncthreads();
        // A tile: 128x64 bf16 (async b128)
#pragma unroll
        for (int j = 0; j < 4; j++) {
            int i = tid + j * 256;
            int r = i >> 3, seg = i & 7;
            ld_g2l_b128(As + r * 64 + seg * 8,
                        A + (size_t)(mBlk + r) * lda + k0 + seg * 8);
        }
        // B tile: 128x64 bf16 from transposed weights
#pragma unroll
        for (int j = 0; j < 4; j++) {
            int i = tid + j * 256;
            int r = i >> 3, seg = i & 7;
            ld_g2l_b128(Bs + r * 64 + seg * 8,
                        Bt + (size_t)(nBlk + r) * ldb + k0 + seg * 8);
        }
        if (k0 + 64 < K) {
            __builtin_prefetch(A + (size_t)(mBlk + (tid >> 1)) * lda + k0 + 64, 0, 0);
            __builtin_prefetch(Bt + (size_t)(nBlk + (tid >> 1)) * ldb + k0 + 64, 0, 0);
        }
        wait_async_loads();
        __syncthreads();

#pragma unroll
        for (int kc = 0; kc < 2; kc++) {
            BF16Frag af[4], bfr[2];
#pragma unroll
            for (int im = 0; im < 4; im++)
                load_frag_rm(af[im], As, 64, mW + im * 16 + (lane & 15), kc * 32, lane);
#pragma unroll
            for (int in = 0; in < 2; in++)
                load_frag_rm(bfr[in], Bs, 64, nW + in * 16 + (lane & 15), kc * 32, lane);
#pragma unroll
            for (int im = 0; im < 4; im++)
#pragma unroll
                for (int in = 0; in < 2; in++)
                    acc[im][in] = __builtin_amdgcn_wmma_f32_16x16x32_bf16(
                        false, af[im].v, false, bfr[in].v, (short)0, acc[im][in], false, false);
        }
    }

    // epilogue: C/D layout — VGPR i: M = i + 8*(lane>=16), N = lane&15
    int rb = (lane >> 4) << 3, cb = lane & 15;
    float bcol[2] = {0.f, 0.f};
    if (bias) {
        bcol[0] = bias[nBlk + nW + cb];
        bcol[1] = bias[nBlk + nW + 16 + cb];
    }
#pragma unroll
    for (int im = 0; im < 4; im++)
#pragma unroll
        for (int in = 0; in < 2; in++)
#pragma unroll
            for (int i = 0; i < 8; i++) {
                int gr = mBlk + mW + im * 16 + rb + i;
                int gc = nBlk + nW + in * 16 + cb;
                float v = acc[im][in][i] + bcol[in];
                if (relu) v = fmaxf(v, 0.f);
                if (rowScale) v *= rowScale[gr];
                if (Cacc) {
                    Cacc[(size_t)gr * ldc + gc] += v;
                } else if (transT) {
                    // token gr = s*B + b  ->  vT[(b*D + gc)*S + s]
                    size_t idx = ((size_t)(gr & (BATCH - 1)) * DMODEL + gc) * S_LEN
                               + (size_t)(gr >> 1);
                    Cbf[idx] = f2bf(v);
                } else {
                    Cbf[(size_t)gr * ldc + gc] = f2bf(v);
                }
            }
}

// ---------------- flash attention (WMMA QK^T + online softmax + WMMA PV) ----------------
// grid: (S/64, H, EA*B), block: 128 (4 waves); each wave owns 16 query rows.
// V comes pre-transposed: vT[(b*D + h*64 + d)*S + s]
__global__ __launch_bounds__(128)
void attn_kernel(const unsigned short* __restrict__ qbf,
                 const unsigned short* __restrict__ kbf,
                 const unsigned short* __restrict__ vT,
                 const unsigned char* __restrict__ pad,   // [B,S]
                 unsigned short* __restrict__ ctx) {
    __shared__ unsigned short Qs[64 * 64];     // [q][d]
    __shared__ unsigned short Ks[64 * 64];     // [key][d]
    __shared__ unsigned short Vs[64 * 64];     // [d][key]  (transposed)
    __shared__ unsigned short Ps[4 * 16 * 64]; // per-wave P staging
    __shared__ float maskAdd[64];

    int tid = threadIdx.x, lane = tid & 31, w = tid >> 5;
    int qt = blockIdx.x, h = blockIdx.y;
    int e = blockIdx.z / BATCH, b = blockIdx.z % BATCH;
    int qbase = qt * 64;
    size_t headOff = (size_t)h * HDIM;
    const unsigned short* qp = qbf + (size_t)e * NTOK * DMODEL;
    const unsigned short* vp = vT + ((size_t)b * DMODEL + headOff) * S_LEN;

    for (int i = tid; i < 512; i += 128) {
        int r = i >> 3, seg = i & 7;
        size_t g = ((size_t)(qbase + r) * BATCH + b) * DMODEL + headOff + seg * 8;
        ld_g2l_b128(Qs + r * 64 + seg * 8, qp + g);
    }
    wait_async_loads();

    float m8[8], l8[8];
    v8f O[4];
#pragma unroll
    for (int i = 0; i < 8; i++) { m8[i] = -3e38f; l8[i] = 0.f; }
#pragma unroll
    for (int t = 0; t < 4; t++)
#pragma unroll
        for (int i = 0; i < 8; i++) O[t][i] = 0.f;

    for (int kt = 0; kt < S_LEN / 64; kt++) {
        __syncthreads();
        int kb0 = kt * 64;
        for (int i = tid; i < 512; i += 128) {
            int r = i >> 3, seg = i & 7;
            size_t gk = ((size_t)(kb0 + r) * BATCH + b) * DMODEL + headOff + seg * 8;
            ld_g2l_b128(Ks + r * 64 + seg * 8, kbf + gk);
            size_t gv = (size_t)r * S_LEN + kb0 + seg * 8;   // r = d index
            ld_g2l_b128(Vs + r * 64 + seg * 8, vp + gv);
        }
        if (tid < 64) maskAdd[tid] = pad[b * S_LEN + kb0 + tid] ? -1e8f : 0.f;
        wait_async_loads();
        __syncthreads();

        // S-tile = Q @ K^T  (M=16 rows of this wave, N=64 keys, K=64 in 2 chunks)
        v8f sacc[4];
#pragma unroll
        for (int nt = 0; nt < 4; nt++)
#pragma unroll
            for (int i = 0; i < 8; i++) sacc[nt][i] = 0.f;
#pragma unroll
        for (int kc = 0; kc < 2; kc++) {
            BF16Frag af;
            load_frag_rm(af, Qs, 64, w * 16 + (lane & 15), kc * 32, lane);
#pragma unroll
            for (int nt = 0; nt < 4; nt++) {
                BF16Frag kf;
                load_frag_rm(kf, Ks, 64, nt * 16 + (lane & 15), kc * 32, lane);
                sacc[nt] = __builtin_amdgcn_wmma_f32_16x16x32_bf16(
                    false, af.v, false, kf.v, (short)0, sacc[nt], false, false);
            }
        }

        // scale + mask
        float sv[4][8];
#pragma unroll
        for (int nt = 0; nt < 4; nt++)
#pragma unroll
            for (int i = 0; i < 8; i++)
                sv[nt][i] = sacc[nt][i] * 0.125f + maskAdd[nt * 16 + (lane & 15)];

        // online softmax (rows live across 16 lanes of each half-wave)
        float scl[8];
#pragma unroll
        for (int i = 0; i < 8; i++) {
            float t = sv[0][i];
#pragma unroll
            for (int nt = 1; nt < 4; nt++) t = fmaxf(t, sv[nt][i]);
#pragma unroll
            for (int o = 8; o >= 1; o >>= 1) t = fmaxf(t, __shfl_xor(t, o, 32));
            float nm = fmaxf(m8[i], t);
            scl[i] = __expf(m8[i] - nm);
            float s = 0.f;
#pragma unroll
            for (int nt = 0; nt < 4; nt++) {
                float p = __expf(sv[nt][i] - nm);
                sv[nt][i] = p;
                s += p;
            }
#pragma unroll
            for (int o = 8; o >= 1; o >>= 1) s += __shfl_xor(s, o, 32);
            l8[i] = l8[i] * scl[i] + s;
            m8[i] = nm;
        }
#pragma unroll
        for (int t = 0; t < 4; t++)
#pragma unroll
            for (int i = 0; i < 8; i++) O[t][i] *= scl[i];

        // stage P (accumulator layout -> row-major bf16 in per-wave LDS)
        int rbase = (lane >> 4) << 3;
#pragma unroll
        for (int nt = 0; nt < 4; nt++)
#pragma unroll
            for (int i = 0; i < 8; i++)
                Ps[w * 1024 + (rbase + i) * 64 + nt * 16 + (lane & 15)] = f2bf(sv[nt][i]);

        // O += P @ V   (V fragment from transposed Vs: row = d, contiguous keys)
#pragma unroll
        for (int kc = 0; kc < 2; kc++) {
            BF16Frag pf;
            load_frag_rm(pf, Ps + w * 1024, 64, (lane & 15), kc * 32, lane);
#pragma unroll
            for (int nt = 0; nt < 4; nt++) {
                BF16Frag vf;
                load_frag_rm(vf, Vs, 64, nt * 16 + (lane & 15), kc * 32, lane);
                O[nt] = __builtin_amdgcn_wmma_f32_16x16x32_bf16(
                    false, pf.v, false, vf.v, (short)0, O[nt], false, false);
            }
        }
    }

    // normalize + write ctx (bf16)
    int rbase = (lane >> 4) << 3, cb = lane & 15;
#pragma unroll
    for (int nt = 0; nt < 4; nt++)
#pragma unroll
        for (int i = 0; i < 8; i++) {
            int srow = qbase + w * 16 + rbase + i;
            float v = O[nt][i] / l8[i];
            size_t g = (size_t)e * NTOK * DMODEL +
                       ((size_t)srow * BATCH + b) * DMODEL + headOff + nt * 16 + cb;
            ctx[g] = f2bf(v);
        }
}

// ---------------- aux loss finalize ----------------
__global__ void aux_finalize_kernel(const float* __restrict__ acc, float* __restrict__ out) {
    // acc: [0..3] impA, [4..7] loadA, [8] zA, [9..16] impF, [17..24] loadF, [25] zF
    auto one = [&](const float* ip, const float* lp, float z, int E) -> float {
        float si = 0.f, si2 = 0.f, sl = 0.f;
        for (int e = 0; e < E; e++) { si += ip[e]; si2 += ip[e] * ip[e]; sl += lp[e]; }
        float mean = si / E;
        float var  = si2 / E - mean * mean;
        float cv   = var / (mean * mean + 1e-10f);
        float sw = 0.f;
        for (int e = 0; e < E; e++)
            sw += (lp[e] / (sl + 1e-9f)) * (ip[e] / (si + 1e-9f));
        sw *= (float)E;
        float zm = z / (float)NTOK;
        return 0.01f * cv + 0.01f * sw + 0.001f * zm;
    };
    out[0] = one(acc, acc + 4, acc[8], EA_N) + one(acc + 9, acc + 17, acc[25], EF_N);
}

// ---------------- host launch ----------------
extern "C" void kernel_launch(void* const* d_in, const int* in_sizes, int n_in,
                              void* d_out, int out_size, void* d_ws, size_t ws_size,
                              hipStream_t stream) {
    (void)in_sizes; (void)n_in; (void)out_size; (void)ws_size;

    const float*         x     = (const float*)d_in[0];
    const unsigned char* halt  = (const unsigned char*)d_in[1];
    // d_in[2] = layer_idx (unused)
    const unsigned char* pad   = (const unsigned char*)d_in[3];
    const float* ln1s = (const float*)d_in[4];
    const float* ln1b = (const float*)d_in[5];
    const float* ln2s = (const float*)d_in[6];
    const float* ln2b = (const float*)d_in[7];
    const float* Wg_a = (const float*)d_in[8];
    const float* Wq   = (const float*)d_in[9];
    const float* bq   = (const float*)d_in[10];
    const float* Wk   = (const float*)d_in[11];
    const float* bk   = (const float*)d_in[12];
    const float* Wv   = (const float*)d_in[13];
    const float* bv   = (const float*)d_in[14];
    const float* Wo   = (const float*)d_in[15];
    const float* bo   = (const float*)d_in[16];
    const float* Wg_f = (const float*)d_in[17];
    const float* W1   = (const float*)d_in[18];
    const float* b1   = (const float*)d_in[19];
    const float* W2   = (const float*)d_in[20];
    const float* b2   = (const float*)d_in[21];

    char* w = (char*)d_ws;
    float* x_mid            = (float*)w;          w += (size_t)NTOK * DMODEL * 4;
    unsigned short* h1      = (unsigned short*)w; w += (size_t)NTOK * DMODEL * 2;
    unsigned short* h2      = (unsigned short*)w; w += (size_t)NTOK * DMODEL * 2;
    unsigned short* kbf     = (unsigned short*)w; w += (size_t)NTOK * DMODEL * 2;
    unsigned short* vTbf    = (unsigned short*)w; w += (size_t)BATCH * DMODEL * S_LEN * 2;
    unsigned short* qbf     = (unsigned short*)w; w += (size_t)EA_N * NTOK * DMODEL * 2;
    unsigned short* ctxbf   = (unsigned short*)w; w += (size_t)EA_N * NTOK * DMODEL * 2;
    unsigned short* hidbf   = (unsigned short*)w; w += (size_t)NTOK * FF_DIM * 2;
    float* gates_a          = (float*)w;          w += (size_t)EA_N * NTOK * 4;
    float* gates_f          = (float*)w;          w += (size_t)EF_N * NTOK * 4;
    float* auxacc           = (float*)w;          w += 32 * 4;
    // transposed bf16 weight mirrors [N,K]
    unsigned short* WqT     = (unsigned short*)w; w += (size_t)EA_N * DMODEL * DMODEL * 2;
    unsigned short* WkT     = (unsigned short*)w; w += (size_t)DMODEL * DMODEL * 2;
    unsigned short* WvT     = (unsigned short*)w; w += (size_t)DMODEL * DMODEL * 2;
    unsigned short* WoT     = (unsigned short*)w; w += (size_t)EA_N * DMODEL * DMODEL * 2;
    unsigned short* W1T     = (unsigned short*)w; w += (size_t)EF_N * DMODEL * FF_DIM * 2;
    unsigned short* W2T     = (unsigned short*)w; w += (size_t)EF_N * FF_DIM * DMODEL * 2;

    const int N4 = NTOK * DMODEL / 4;
    dim3 gemm8(DMODEL / 128, NTOK / 128);    // N=1024 outputs
    dim3 gemm16(FF_DIM / 128, NTOK / 128);   // N=2048 outputs
    dim3 tDD(DMODEL / 64, DMODEL / 64);      // transpose grids
    dim3 tDF(DMODEL / 64, FF_DIM / 64);
    dim3 tFD(FF_DIM / 64, DMODEL / 64);

    // one-shot weight conversion: f32 [K,N] -> bf16 transposed [N,K]
    convT_bf16_kernel<<<tDD, 256, 0, stream>>>(Wk, WkT, DMODEL, DMODEL);
    convT_bf16_kernel<<<tDD, 256, 0, stream>>>(Wv, WvT, DMODEL, DMODEL);
    for (int e = 0; e < EA_N; e++) {
        convT_bf16_kernel<<<tDD, 256, 0, stream>>>(Wq + (size_t)e * DMODEL * DMODEL,
                                                   WqT + (size_t)e * DMODEL * DMODEL,
                                                   DMODEL, DMODEL);
        convT_bf16_kernel<<<tDD, 256, 0, stream>>>(Wo + (size_t)e * DMODEL * DMODEL,
                                                   WoT + (size_t)e * DMODEL * DMODEL,
                                                   DMODEL, DMODEL);
    }
    for (int e = 0; e < EF_N; e++) {
        convT_bf16_kernel<<<tDF, 256, 0, stream>>>(W1 + (size_t)e * DMODEL * FF_DIM,
                                                   W1T + (size_t)e * DMODEL * FF_DIM,
                                                   DMODEL, FF_DIM);
        convT_bf16_kernel<<<tFD, 256, 0, stream>>>(W2 + (size_t)e * FF_DIM * DMODEL,
                                                   W2T + (size_t)e * FF_DIM * DMODEL,
                                                   FF_DIM, DMODEL);
    }

    // residual buffer = x
    copy_f4_kernel<<<(N4 + 255) / 256, 256, 0, stream>>>((const float4*)x, (float4*)x_mid, N4);
    zero_aux_kernel<<<1, 32, 0, stream>>>(auxacc);

    // ---- attention block ----
    ln_bf16_kernel<<<NTOK, 256, 0, stream>>>(x, ln1s, ln1b, h1);
    gate_kernel<EA_N><<<NTOK / 8, 256, 0, stream>>>(h1, Wg_a, halt, gates_a,
                                                    auxacc, auxacc + 4, auxacc + 8);
    gemm_bf16_kernel<<<gemm8, 256, 0, stream>>>(h1, DMODEL, WkT, DMODEL, bk, nullptr,
                                                kbf, nullptr, DMODEL, DMODEL, 0, 0);
    gemm_bf16_kernel<<<gemm8, 256, 0, stream>>>(h1, DMODEL, WvT, DMODEL, bv, nullptr,
                                                vTbf, nullptr, DMODEL, DMODEL, 0, 1);
    for (int e = 0; e < EA_N; e++) {
        gemm_bf16_kernel<<<gemm8, 256, 0, stream>>>(
            h1, DMODEL, WqT + (size_t)e * DMODEL * DMODEL, DMODEL, bq + (size_t)e * DMODEL,
            nullptr, qbf + (size_t)e * NTOK * DMODEL, nullptr, DMODEL, DMODEL, 0, 0);
    }
    attn_kernel<<<dim3(S_LEN / 64, NHEAD, EA_N * BATCH), 128, 0, stream>>>(
        qbf, kbf, vTbf, pad, ctxbf);
    for (int e = 0; e < EA_N; e++) {
        gemm_bf16_kernel<<<gemm8, 256, 0, stream>>>(
            ctxbf + (size_t)e * NTOK * DMODEL, DMODEL,
            WoT + (size_t)e * DMODEL * DMODEL, DMODEL, bo + (size_t)e * DMODEL,
            gates_a + (size_t)e * NTOK, nullptr, x_mid, DMODEL, DMODEL, 0, 0);
    }

    // ---- FFN block ----
    ln_bf16_kernel<<<NTOK, 256, 0, stream>>>(x_mid, ln2s, ln2b, h2);
    gate_kernel<EF_N><<<NTOK / 8, 256, 0, stream>>>(h2, Wg_f, halt, gates_f,
                                                    auxacc + 9, auxacc + 17, auxacc + 25);
    for (int e = 0; e < EF_N; e++) {
        gemm_bf16_kernel<<<gemm16, 256, 0, stream>>>(
            h2, DMODEL, W1T + (size_t)e * DMODEL * FF_DIM, DMODEL, b1 + (size_t)e * FF_DIM,
            nullptr, hidbf, nullptr, FF_DIM, DMODEL, 1, 0);
        gemm_bf16_kernel<<<gemm8, 256, 0, stream>>>(
            hidbf, FF_DIM, W2T + (size_t)e * FF_DIM * DMODEL, FF_DIM, b2 + (size_t)e * DMODEL,
            gates_f + (size_t)e * NTOK, nullptr, x_mid, DMODEL, FF_DIM, 0, 0);
    }

    // ---- outputs ----
    copy_f4_kernel<<<(N4 + 255) / 256, 256, 0, stream>>>((const float4*)x_mid, (float4*)d_out, N4);
    aux_finalize_kernel<<<1, 1, 0, stream>>>(auxacc, (float*)d_out + (size_t)NTOK * DMODEL);
}
